// RPLNet_66563403153701
// MI455X (gfx1250) — matrix operand order repaired and checked by reading, not compiled, not run on hardware
//
#include <hip/hip_runtime.h>

// ---------------------------------------------------------------------------
// Problem constants (match reference: B=32, N=2048, D=3)
// ---------------------------------------------------------------------------
#define BATCH   32
#define NPTS    2048
#define ROWS    (BATCH * NPTS)      // 65536
#define MLAT    (NPTS * 4)          // 8192 max unique lattice points / cloud
#define MROWS   (MLAT + 1)          // +1 zero row for missing neighbors
#define NTAPS   9
#define HBITS   15
#define HSIZE   (1 << HBITS)        // 32768 slots, <=0.25 load factor
#define HMASK   (HSIZE - 1)
#define KEMPTY  0xFFFFFFFFFFFFFFFFull

typedef __attribute__((ext_vector_type(16))) __bf16 v16bf;
typedef __attribute__((ext_vector_type(8)))  __bf16 v8bf;
typedef __attribute__((ext_vector_type(8)))  float  v8f;

__device__ __constant__ int NEIGH_T[NTAPS][3] = {
    { 0, 0, 0},
    { 3,-1,-1}, {-3, 1, 1},
    {-1, 3,-1}, { 1,-3, 1},
    {-1,-1, 3}, { 1, 1,-3},
    {-1,-1,-1}, { 1, 1, 1}
};

__device__ __forceinline__ unsigned long long enc3(int k0, int k1, int k2) {
    unsigned long long a = (unsigned long long)(k0 + 131072);
    unsigned long long b = (unsigned long long)(k1 + 131072);
    unsigned long long c = (unsigned long long)(k2 + 131072);
    return (a * 262144ull + b) * 262144ull + c;
}

__device__ __forceinline__ unsigned int hash64(unsigned long long x) {
    x ^= x >> 33; x *= 0xff51afd7ed558ccdULL;
    x ^= x >> 33; x *= 0xc4ceb9fe1a85ec53ULL;
    x ^= x >> 33;
    return (unsigned int)x;
}

// ---------------------------------------------------------------------------
// Fills
// ---------------------------------------------------------------------------
__global__ void k_fill_f32(float* p, float v, size_t n) {
    size_t i = blockIdx.x * (size_t)blockDim.x + threadIdx.x;
    if (i < n) p[i] = v;
}
__global__ void k_fill_i32(int* p, int v, size_t n) {
    size_t i = blockIdx.x * (size_t)blockDim.x + threadIdx.x;
    if (i < n) p[i] = v;
}
__global__ void k_fill_u64(unsigned long long* p, unsigned long long v, size_t n) {
    size_t i = blockIdx.x * (size_t)blockDim.x + threadIdx.x;
    if (i < n) p[i] = v;
}

// ---------------------------------------------------------------------------
// f32 -> bf16 conversion with optional K padding (A panels, weights, filters)
// ---------------------------------------------------------------------------
__global__ void k_cvt_pad(const float* __restrict__ src, int ld, int K,
                          __bf16* __restrict__ dst, int Kp, int rows) {
    size_t i = blockIdx.x * (size_t)blockDim.x + threadIdx.x;
    if (i >= (size_t)rows * Kp) return;
    int k = (int)(i % Kp);
    size_t r = i / Kp;
    float v = (k < K) ? src[r * ld + k] : 0.f;
    dst[i] = (__bf16)v;
}

// vals f32 [B][MLAT][Ci] -> bf16 [B][MROWS][Ci], row MLAT zeroed
__global__ void k_cvt_vals(const float* __restrict__ vals, __bf16* __restrict__ out, int Ci) {
    size_t i = blockIdx.x * (size_t)blockDim.x + threadIdx.x;
    if (i >= (size_t)BATCH * MROWS * Ci) return;
    int c = (int)(i % Ci);
    size_t rc = i / Ci;
    int m = (int)(rc % MROWS);
    int b = (int)(rc / MROWS);
    float v = (m < MLAT) ? vals[((size_t)b * MLAT + m) * Ci + c] : 0.f;
    out[i] = (__bf16)v;
}

// ---------------------------------------------------------------------------
// Layer 1: 3 -> 32 tiny dense
// ---------------------------------------------------------------------------
__global__ void k_linear1(const float* __restrict__ pc, const float* __restrict__ w,
                          const float* __restrict__ bias, float* __restrict__ out) {
    size_t i = blockIdx.x * (size_t)blockDim.x + threadIdx.x;
    if (i >= (size_t)ROWS * 32) return;
    int c = (int)(i & 31);
    size_t row = i >> 5;
    const float* p = pc + row * 3;
    const float* wr = w + c * 3;
    out[i] = p[0] * wr[0] + p[1] * wr[1] + p[2] * wr[2] + bias[c];
}

// ---------------------------------------------------------------------------
// BatchNorm: per-channel block reduction, then fused normalize + ReLU
// ---------------------------------------------------------------------------
__global__ void k_bn_stats(const float* __restrict__ x, int ld,
                           float* __restrict__ sum, float* __restrict__ sq) {
    int c = blockIdx.x;
    float s = 0.f, s2 = 0.f;
    for (int r = threadIdx.x; r < ROWS; r += blockDim.x) {
        float v = x[(size_t)r * ld + c];
        s += v; s2 += v * v;
    }
    __shared__ float sh[512];
    sh[threadIdx.x] = s;
    sh[256 + threadIdx.x] = s2;
    __syncthreads();
    for (int off = 128; off > 0; off >>= 1) {
        if ((int)threadIdx.x < off) {
            sh[threadIdx.x] += sh[threadIdx.x + off];
            sh[256 + threadIdx.x] += sh[256 + threadIdx.x + off];
        }
        __syncthreads();
    }
    if (threadIdx.x == 0) { sum[c] = sh[0]; sq[c] = sh[256]; }
}

__global__ void k_bn_apply(float* __restrict__ x, int ld, int C,
                           const float* __restrict__ sum, const float* __restrict__ sq,
                           const float* __restrict__ g, const float* __restrict__ be) {
    size_t i = blockIdx.x * (size_t)blockDim.x + threadIdx.x;
    if (i >= (size_t)ROWS * C) return;
    int c = (int)(i % C);
    size_t r = i / C;
    float inv = 1.0f / (float)ROWS;
    float mu = sum[c] * inv;
    float var = sq[c] * inv - mu * mu;
    float v = x[r * ld + c];
    v = (v - mu) * rsqrtf(var + 1e-5f) * g[c] + be[c];
    x[r * ld + c] = fmaxf(v, 0.f);
}

// ---------------------------------------------------------------------------
// Permutohedral lattice: barycentric coords + hash-insert unique keys
// ---------------------------------------------------------------------------
__global__ void k_lattice(const float* __restrict__ pc, float scale,
                          float* __restrict__ wbary, int* __restrict__ lat,
                          int* __restrict__ ukeys,
                          unsigned long long* __restrict__ hkeys,
                          int* __restrict__ hvals, int* __restrict__ hcount) {
    int t = blockIdx.x * blockDim.x + threadIdx.x;
    if (t >= ROWS) return;
    int b = t / NPTS;

    const float rt23 = 0.81649658092772603f;  // sqrt(2/3)
    float cf[3];
    #pragma unroll
    for (int j = 0; j < 3; ++j) {
        float sj = 4.0f * rt23 * rsqrtf((float)((j + 1) * (j + 2)));
        cf[j] = pc[(size_t)t * 3 + j] * scale * sj;
    }
    float suff[4];
    suff[3] = 0.f; suff[2] = cf[2]; suff[1] = cf[1] + cf[2]; suff[0] = cf[0] + cf[1] + cf[2];
    float elev[4];
    elev[0] = suff[0];
    elev[1] = suff[1] - 1.f * cf[0];
    elev[2] = suff[2] - 2.f * cf[1];
    elev[3] = suff[3] - 3.f * cf[2];

    float rem0[4], diff[4];
    #pragma unroll
    for (int k = 0; k < 4; ++k) {
        float v = elev[k] * 0.25f;
        float up = ceilf(v) * 4.f, dn = floorf(v) * 4.f;
        rem0[k] = (up - elev[k] < elev[k] - dn) ? up : dn;
        diff[k] = elev[k] - rem0[k];
    }
    int rank[4];
    #pragma unroll
    for (int a = 0; a < 4; ++a) {
        int r = 0;
        #pragma unroll
        for (int bb = 0; bb < 4; ++bb)
            r += (diff[bb] > diff[a]) || ((diff[bb] == diff[a]) && (bb < a));
        rank[a] = r;
    }
    int s = (int)rintf((rem0[0] + rem0[1] + rem0[2] + rem0[3]) * 0.25f);
    #pragma unroll
    for (int k = 0; k < 4; ++k) {
        rank[k] += s;
        if (rank[k] < 0)      { rank[k] += 4; rem0[k] += 4.f; }
        else if (rank[k] > 3) { rank[k] -= 4; rem0[k] -= 4.f; }
    }
    float bary[5] = {0.f, 0.f, 0.f, 0.f, 0.f};
    #pragma unroll
    for (int k = 0; k < 4; ++k) {
        float vv = (elev[k] - rem0[k]) * 0.25f;
        bary[3 - rank[k]] += vv;
        bary[4 - rank[k]] -= vv;
    }
    bary[0] += 1.0f + bary[4];

    int r0i[3];
    #pragma unroll
    for (int j = 0; j < 3; ++j) r0i[j] = (int)rintf(rem0[j]);

    unsigned long long* hk = hkeys + (size_t)b * HSIZE;
    int* hv = hvals + (size_t)b * HSIZE;

    #pragma unroll
    for (int r = 0; r < 4; ++r) {
        int key[3];
        #pragma unroll
        for (int j = 0; j < 3; ++j)
            key[j] = r0i[j] + r - 4 * (rank[j] > 3 - r);
        unsigned long long code = enc3(key[0], key[1], key[2]);
        unsigned int slot = hash64(code) & HMASK;
        int uid = -1;
        for (;;) {
            unsigned long long old = atomicCAS(&hk[slot], KEMPTY, code);
            if (old == KEMPTY) {
                uid = atomicAdd(&hcount[b], 1);
                int* uk = ukeys + ((size_t)b * MLAT + uid) * 3;
                uk[0] = key[0]; uk[1] = key[1]; uk[2] = key[2];
                __threadfence();
                atomicExch(&hv[slot], uid);
                break;
            } else if (old == code) {
                volatile int* p = &hv[slot];
                int u;
                while ((u = *p) < 0) { }
                uid = u;
                break;
            } else {
                slot = (slot + 1) & HMASK;
            }
        }
        lat[(size_t)t * 4 + r]   = uid;
        wbary[(size_t)t * 4 + r] = bary[r];
    }
}

// ---------------------------------------------------------------------------
// Neighbor indices for the 9 blur taps (hash lookup; MLAT = zero row if absent)
// ---------------------------------------------------------------------------
__global__ void k_neighbors(const int* __restrict__ ukeys, const int* __restrict__ hcount,
                            const unsigned long long* __restrict__ hkeys,
                            const int* __restrict__ hvals, int* __restrict__ nidx) {
    int t = blockIdx.x * blockDim.x + threadIdx.x;
    if (t >= BATCH * MLAT) return;
    int b = t / MLAT, m = t % MLAT;
    int cnt = hcount[b];
    if (m >= cnt) {
        for (int tap = 0; tap < NTAPS; ++tap)
            nidx[((size_t)b * NTAPS + tap) * MLAT + m] = MLAT;   // zero row
        return;
    }
    const int* uk = ukeys + (size_t)t * 3;
    const unsigned long long* hk = hkeys + (size_t)b * HSIZE;
    const int* hv = hvals + (size_t)b * HSIZE;
    for (int tap = 0; tap < NTAPS; ++tap) {
        unsigned long long code = enc3(uk[0] + NEIGH_T[tap][0],
                                       uk[1] + NEIGH_T[tap][1],
                                       uk[2] + NEIGH_T[tap][2]);
        unsigned int slot = hash64(code) & HMASK;
        int res = MLAT;                                          // zero row
        for (int probe = 0; probe < HSIZE; ++probe) {
            unsigned long long kk = hk[slot];
            if (kk == KEMPTY) break;
            if (kk == code) { res = hv[slot]; break; }
            slot = (slot + 1) & HMASK;
        }
        nidx[((size_t)b * NTAPS + tap) * MLAT + m] = res;
    }
}

// ---------------------------------------------------------------------------
// Splat: vals[lat] += w * feat  (f32 atomics)
// ---------------------------------------------------------------------------
__global__ void k_splat(const float* __restrict__ x, int ld, int Ci,
                        const float* __restrict__ wbary, const int* __restrict__ lat,
                        float* __restrict__ vals) {
    size_t i = blockIdx.x * (size_t)blockDim.x + threadIdx.x;
    if (i >= (size_t)ROWS * 4) return;
    size_t row = i >> 2;                  // b*N + n
    int b = (int)(row / NPTS);
    float wv = wbary[i];
    int u = lat[i];
    const float* xr = x + row * ld;
    float* vr = vals + ((size_t)b * MLAT + u) * Ci;
    for (int c = 0; c < Ci; ++c)
        atomicAdd(&vr[c], wv * xr[c]);
}

// ---------------------------------------------------------------------------
// Blur: blurred[m,:] = sum_t vals[nidx[t][m],:] @ filt[t].T   (WMMA bf16)
// One wave computes a 16x64 output tile (4 accumulators): each A fragment is
// loaded once per K-chunk and reused by 4 WMMAs -> 4x less A traffic.
// Co in {64,128,256} (multiple of 64): no masking anywhere.
// ---------------------------------------------------------------------------
__global__ void k_blur_wmma(const __bf16* __restrict__ valsbf, const __bf16* __restrict__ filtbf,
                            const int* __restrict__ nidx, const int* __restrict__ hcount,
                            float* __restrict__ blurred, int Ci, int Co) {
    int wave = threadIdx.x >> 5;
    int lane = threadIdx.x & 31;
    int b  = blockIdx.z;
    int m0 = (blockIdx.x * 4 + wave) * 16;
    int cnt = hcount[b];
    if (m0 >= cnt) return;                 // uniform within wave

    int half = lane >> 4;
    int l16  = lane & 15;
    int arow = m0 + l16;                   // A row this lane feeds
    int colbase = blockIdx.y * 64 + l16;   // first of 4 column sub-tiles
    size_t fstep = (size_t)16 * Ci;        // 16 filter rows

    v8f acc[4];
    #pragma unroll
    for (int j = 0; j < 4; ++j) acc[j] = (v8f){0.f,0.f,0.f,0.f,0.f,0.f,0.f,0.f};

    for (int t = 0; t < NTAPS; ++t) {
        int src = nidx[((size_t)b * NTAPS + t) * MLAT + arow];      // in [0, MLAT]
        const __bf16* ap = valsbf + ((size_t)b * MROWS + src) * Ci + 16 * half;
        const __bf16* fp = filtbf + ((size_t)t * Co + colbase) * Ci + 16 * half;
        for (int k0 = 0; k0 < Ci; k0 += 32) {
            v8bf lo = *(const v8bf*)(ap + k0);
            v8bf hi = *(const v8bf*)(ap + k0 + 8);
            v16bf af = __builtin_shufflevector(lo, hi,
                0, 1, 2, 3, 4, 5, 6, 7, 8, 9, 10, 11, 12, 13, 14, 15);
            #pragma unroll
            for (int j = 0; j < 4; ++j) {
                v16bf bfm = *(const v16bf*)(fp + (size_t)j * fstep + k0);
                acc[j] = __builtin_amdgcn_wmma_f32_16x16x32_bf16(
                    false, af, false, bfm, (short)0, acc[j], false, false);
            }
        }
    }
    // D layout: VGPR v -> row m0 + v + half*8, col = colbase + j*16
    #pragma unroll
    for (int j = 0; j < 4; ++j) {
        #pragma unroll
        for (int v = 0; v < 8; ++v) {
            int row = m0 + v + half * 8;
            blurred[((size_t)b * MLAT + row) * Co + colbase + j * 16] = acc[j][v];
        }
    }
}

// ---------------------------------------------------------------------------
// Slice: out[n,c] = sum_r w[n,r] * blurred[lat[n,r], c] + bias[c]
// ---------------------------------------------------------------------------
__global__ void k_slice(const float* __restrict__ blurred, const float* __restrict__ wbary,
                        const int* __restrict__ lat, const float* __restrict__ pb,
                        float* __restrict__ out, int ld, int Co) {
    size_t i = blockIdx.x * (size_t)blockDim.x + threadIdx.x;
    if (i >= (size_t)ROWS * Co) return;
    int c = (int)(i % Co);
    size_t row = i / Co;
    int b = (int)(row / NPTS);
    float acc = pb[c];
    #pragma unroll
    for (int r = 0; r < 4; ++r) {
        int u = lat[row * 4 + r];
        acc += wbary[row * 4 + r] * blurred[((size_t)b * MLAT + u) * Co + c];
    }
    out[row * ld + c] = acc;
}

// ---------------------------------------------------------------------------
// Tail GEMM: C[m,co] = sum_k A[m,k]*W[co,k] + bias[co]   (WMMA bf16, padded K)
// 16x64 tile per wave (4 accs), A fragment reused by 4 WMMAs per K-chunk.
// Loads clamp the column; only stores mask Co edges.
// ---------------------------------------------------------------------------
__global__ void k_gemm_wmma(const __bf16* __restrict__ A, int Kp,
                            const __bf16* __restrict__ W, const float* __restrict__ bias,
                            float* __restrict__ Cmat, int ldc, int Co) {
    int wave = threadIdx.x >> 5;
    int lane = threadIdx.x & 31;
    int m0 = (blockIdx.x * 4 + wave) * 16;
    int half = lane >> 4;
    int l16  = lane & 15;
    int colbase = blockIdx.y * 64 + l16;

    const __bf16* arow = A + (size_t)(m0 + l16) * Kp + 16 * half;
    const __bf16* wrow[4];
    #pragma unroll
    for (int j = 0; j < 4; ++j) {
        int col = colbase + j * 16;
        int colc = (col < Co) ? col : (Co - 1);
        wrow[j] = W + (size_t)colc * Kp + 16 * half;
    }

    v8f acc[4];
    #pragma unroll
    for (int j = 0; j < 4; ++j) acc[j] = (v8f){0.f,0.f,0.f,0.f,0.f,0.f,0.f,0.f};

    for (int k0 = 0; k0 < Kp; k0 += 32) {
        v8bf lo = *(const v8bf*)(arow + k0);
        v8bf hi = *(const v8bf*)(arow + k0 + 8);
        v16bf af = __builtin_shufflevector(lo, hi,
            0, 1, 2, 3, 4, 5, 6, 7, 8, 9, 10, 11, 12, 13, 14, 15);
        #pragma unroll
        for (int j = 0; j < 4; ++j) {
            v16bf bfm = *(const v16bf*)(wrow[j] + k0);
            acc[j] = __builtin_amdgcn_wmma_f32_16x16x32_bf16(
                false, af, false, bfm, (short)0, acc[j], false, false);
        }
    }
    #pragma unroll
    for (int j = 0; j < 4; ++j) {
        int col = colbase + j * 16;
        if (col < Co) {
            float bv = bias[col];
            #pragma unroll
            for (int v = 0; v < 8; ++v) {
                int row = m0 + v + half * 8;
                Cmat[(size_t)row * ldc + col] = acc[j][v] + bv;
            }
        }
    }
}

// ---------------------------------------------------------------------------
// Final transpose [B,N,3] -> [B,3,N]
// ---------------------------------------------------------------------------
__global__ void k_transpose3(const float* __restrict__ x3, float* __restrict__ out) {
    size_t i = blockIdx.x * (size_t)blockDim.x + threadIdx.x;
    if (i >= (size_t)ROWS * 3) return;
    int c = (int)(i % 3);
    size_t row = i / 3;
    int n = (int)(row % NPTS);
    int b = (int)(row / NPTS);
    out[((size_t)b * 3 + c) * NPTS + n] = x3[i];
}

// ---------------------------------------------------------------------------
// Host orchestration
// ---------------------------------------------------------------------------
extern "C" void kernel_launch(void* const* d_in, const int* in_sizes, int n_in,
                              void* d_out, int out_size, void* d_ws, size_t ws_size,
                              hipStream_t stream) {
    (void)in_sizes; (void)n_in; (void)out_size; (void)ws_size;

    const float* pc  = (const float*)d_in[0];
    const float* w1  = (const float*)d_in[1];
    const float* b1  = (const float*)d_in[2];
    const float* g1  = (const float*)d_in[3];
    const float* be1 = (const float*)d_in[4];
    const float *filt[5], *pb[5], *gp[5], *bep[5];
    for (int i = 0; i < 5; ++i) {
        filt[i] = (const float*)d_in[5 + 4 * i + 0];
        pb[i]   = (const float*)d_in[5 + 4 * i + 1];
        gp[i]   = (const float*)d_in[5 + 4 * i + 2];
        bep[i]  = (const float*)d_in[5 + 4 * i + 3];
    }
    const float* w7  = (const float*)d_in[25];
    const float* b7  = (const float*)d_in[26];
    const float* g7  = (const float*)d_in[27];
    const float* be7 = (const float*)d_in[28];
    const float* w8  = (const float*)d_in[29];
    const float* b8  = (const float*)d_in[30];
    const float* g8  = (const float*)d_in[31];
    const float* be8 = (const float*)d_in[32];
    const float* w9  = (const float*)d_in[33];
    const float* b9  = (const float*)d_in[34];
    const float* g9  = (const float*)d_in[35];
    const float* be9 = (const float*)d_in[36];

    // Carve workspace
    size_t off = 0;
    auto carve = [&](size_t bytes) -> void* {
        void* p = (char*)d_ws + off;
        off += (bytes + 255) & ~(size_t)255;
        return p;
    };
    float*  f1     = (float*) carve((size_t)ROWS * 32 * 4);
    float*  X960   = (float*) carve((size_t)ROWS * 960 * 4);
    float*  X240   = (float*) carve((size_t)ROWS * 240 * 4);
    float*  X60    = (float*) carve((size_t)ROWS * 60 * 4);
    float*  X3     = (float*) carve((size_t)ROWS * 3 * 4);
    float*  wbary  = (float*) carve((size_t)ROWS * 4 * 4);
    int*    lat    = (int*)   carve((size_t)ROWS * 4 * 4);
    int*    ukeys  = (int*)   carve((size_t)BATCH * MLAT * 3 * 4);
    int*    nidx   = (int*)   carve((size_t)BATCH * NTAPS * MLAT * 4);
    float*  vals   = (float*) carve((size_t)BATCH * MLAT * 256 * 4);
    float*  blur   = (float*) carve((size_t)BATCH * MLAT * 256 * 4);
    __bf16* valsbf = (__bf16*)carve((size_t)BATCH * MROWS * 256 * 2);
    __bf16* filtbf = (__bf16*)carve((size_t)NTAPS * 256 * 256 * 2);
    __bf16* a960bf = (__bf16*)carve((size_t)ROWS * 960 * 2);
    __bf16* a256bf = (__bf16*)carve((size_t)ROWS * 256 * 2);
    __bf16* a64bf  = (__bf16*)carve((size_t)ROWS * 64 * 2);
    __bf16* w7bf   = (__bf16*)carve((size_t)240 * 960 * 2);
    __bf16* w8bf   = (__bf16*)carve((size_t)60 * 256 * 2);
    __bf16* w9bf   = (__bf16*)carve((size_t)3 * 64 * 2);
    unsigned long long* hkeys = (unsigned long long*)carve((size_t)BATCH * HSIZE * 8);
    int*    hvals  = (int*)   carve((size_t)BATCH * HSIZE * 4);
    int*    hcnt   = (int*)   carve((size_t)BATCH * 4);
    float*  bnsum  = (float*) carve(1024 * 4);
    float*  bnsq   = (float*) carve(1024 * 4);

    auto cdiv = [](size_t a, size_t b) -> unsigned { return (unsigned)((a + b - 1) / b); };

    // ---- Layer 1 + BN/ReLU ----
    k_linear1<<<cdiv((size_t)ROWS * 32, 256), 256, 0, stream>>>(pc, w1, b1, f1);
    k_bn_stats<<<32, 256, 0, stream>>>(f1, 32, bnsum, bnsq);
    k_bn_apply<<<cdiv((size_t)ROWS * 32, 256), 256, 0, stream>>>(f1, 32, 32, bnsum, bnsq, g1, be1);

    // ---- Permutohedral layers ----
    struct PL { const float* in; int in_ld; int Ci, Co; float scale; int out_off; int pi; };
    PL cfgs[5] = {
        { f1,         32,  32,  64,  64.f, 0,   0 },
        { X960 + 0,   960, 64,  128, 32.f, 64,  1 },
        { X960 + 64,  960, 128, 256, 16.f, 192, 2 },
        { X960 + 192, 960, 256, 256, 8.f,  448, 3 },
        { X960 + 448, 960, 256, 256, 4.f,  704, 4 },
    };
    for (int li = 0; li < 5; ++li) {
        const PL& c = cfgs[li];
        k_fill_u64<<<cdiv((size_t)BATCH * HSIZE, 256), 256, 0, stream>>>(hkeys, KEMPTY, (size_t)BATCH * HSIZE);
        k_fill_i32<<<cdiv((size_t)BATCH * HSIZE, 256), 256, 0, stream>>>(hvals, -1, (size_t)BATCH * HSIZE);
        k_fill_i32<<<1, 64, 0, stream>>>(hcnt, 0, BATCH);
        k_lattice<<<cdiv(ROWS, 256), 256, 0, stream>>>(pc, c.scale, wbary, lat, ukeys, hkeys, hvals, hcnt);
        k_neighbors<<<cdiv((size_t)BATCH * MLAT, 256), 256, 0, stream>>>(ukeys, hcnt, hkeys, hvals, nidx);
        k_fill_f32<<<cdiv((size_t)BATCH * MLAT * c.Ci, 256), 256, 0, stream>>>(vals, 0.f, (size_t)BATCH * MLAT * c.Ci);
        k_splat<<<cdiv((size_t)ROWS * 4, 256), 256, 0, stream>>>(c.in, c.in_ld, c.Ci, wbary, lat, vals);
        // bf16 operand prep
        k_cvt_vals<<<cdiv((size_t)BATCH * MROWS * c.Ci, 256), 256, 0, stream>>>(vals, valsbf, c.Ci);
        k_cvt_pad<<<cdiv((size_t)NTAPS * c.Co * c.Ci, 256), 256, 0, stream>>>(
            filt[c.pi], c.Ci, c.Ci, filtbf, c.Ci, NTAPS * c.Co);
        dim3 bg(MLAT / 64, c.Co / 64, BATCH);   // 4 waves/block over M; 16x64/wave
        k_blur_wmma<<<bg, 128, 0, stream>>>(valsbf, filtbf, nidx, hcnt, blur, c.Ci, c.Co);
        float* outp = X960 + c.out_off;
        k_slice<<<cdiv((size_t)ROWS * c.Co, 256), 256, 0, stream>>>(blur, wbary, lat, pb[c.pi], outp, 960, c.Co);
        k_bn_stats<<<c.Co, 256, 0, stream>>>(outp, 960, bnsum, bnsq);
        k_bn_apply<<<cdiv((size_t)ROWS * c.Co, 256), 256, 0, stream>>>(outp, 960, c.Co, bnsum, bnsq, gp[c.pi], bep[c.pi]);
    }

    // ---- Tail dense layers (WMMA GEMMs, padded-K bf16 operands) ----
    {
        k_cvt_pad<<<cdiv((size_t)ROWS * 960, 256), 256, 0, stream>>>(X960, 960, 960, a960bf, 960, ROWS);
        k_cvt_pad<<<cdiv((size_t)240 * 960, 256), 256, 0, stream>>>(w7, 960, 960, w7bf, 960, 240);
        dim3 g(ROWS / 64, cdiv(240, 64));
        k_gemm_wmma<<<g, 128, 0, stream>>>(a960bf, 960, w7bf, b7, X240, 240, 240);
        k_bn_stats<<<240, 256, 0, stream>>>(X240, 240, bnsum, bnsq);
        k_bn_apply<<<cdiv((size_t)ROWS * 240, 256), 256, 0, stream>>>(X240, 240, 240, bnsum, bnsq, g7, be7);
    }
    {
        k_cvt_pad<<<cdiv((size_t)ROWS * 256, 256), 256, 0, stream>>>(X240, 240, 240, a256bf, 256, ROWS);
        k_cvt_pad<<<cdiv((size_t)60 * 256, 256), 256, 0, stream>>>(w8, 240, 240, w8bf, 256, 60);
        dim3 g(ROWS / 64, 1);
        k_gemm_wmma<<<g, 128, 0, stream>>>(a256bf, 256, w8bf, b8, X60, 60, 60);
        k_bn_stats<<<60, 256, 0, stream>>>(X60, 60, bnsum, bnsq);
        k_bn_apply<<<cdiv((size_t)ROWS * 60, 256), 256, 0, stream>>>(X60, 60, 60, bnsum, bnsq, g8, be8);
    }
    {
        k_cvt_pad<<<cdiv((size_t)ROWS * 64, 256), 256, 0, stream>>>(X60, 60, 60, a64bf, 64, ROWS);
        k_cvt_pad<<<cdiv((size_t)3 * 64, 64), 64, 0, stream>>>(w9, 60, 60, w9bf, 64, 3);
        dim3 g(ROWS / 64, 1);
        k_gemm_wmma<<<g, 128, 0, stream>>>(a64bf, 64, w9bf, b9, X3, 3, 3);
        k_bn_stats<<<3, 256, 0, stream>>>(X3, 3, bnsum, bnsq);
        k_bn_apply<<<cdiv((size_t)ROWS * 3, 256), 256, 0, stream>>>(X3, 3, 3, bnsum, bnsq, g9, be9);
    }
    k_transpose3<<<cdiv((size_t)ROWS * 3, 256), 256, 0, stream>>>(X3, (float*)d_out);
}
